// NlpClassModel_70806830842334
// MI455X (gfx1250) — matrix operand order repaired and checked by reading, not compiled, not run on hardware
//
#include <hip/hip_runtime.h>
#include <hip/hip_bf16.h>

typedef __attribute__((ext_vector_type(16))) __bf16          v16bf;
typedef __attribute__((ext_vector_type(8)))  float           v8f;
typedef __attribute__((ext_vector_type(8)))  unsigned short  v8us;
typedef __attribute__((ext_vector_type(16))) unsigned short  v16us;

#define VOCAB 29
#define VEC   512
#define HID   1024
#define SEN   64
#define BATCH 2048
#define BT    16          // batch rows per block
#define NCLS  (SEN + 1)   // 65
#define NCLSP 80          // padded to 5 full 16-wide tiles

// ---- helpers -------------------------------------------------------------

__device__ __forceinline__ unsigned short f2b(float f) {
  unsigned int u = __float_as_uint(f);
  u += 0x7FFFu + ((u >> 16) & 1u);   // round-to-nearest-even
  return (unsigned short)(u >> 16);
}

__device__ __forceinline__ unsigned int f2b2(float lo, float hi) {
  return (unsigned int)f2b(lo) | ((unsigned int)f2b(hi) << 16);
}

// Hardware transcendental tanh on CDNA5 (V_TANH_F32, TRANS pipe — co-executes
// with XDL WMMA). Guarded so compilation never regresses.
__device__ __forceinline__ float fast_tanh(float x) {
#if __has_builtin(__builtin_amdgcn_tanhf)
  return __builtin_amdgcn_tanhf(x);
#elif __has_builtin(__builtin_amdgcn_tanh_f32)
  return __builtin_amdgcn_tanh_f32(x);
#else
  return tanhf(x);
#endif
}

// B fragment (32x16 bf16): per lane 16 contiguous K values of one weight row.
__device__ __forceinline__ v16bf load_b_frag(const unsigned short* p) {
  union { v16us u; v16bf b; } t;
  t.u = *(const v16us*)p;            // 32B -> 2x b128
  return t.b;
}

// A fragment (16x32 bf16): per lane 16 contiguous K values as two consecutive 8-element (16B) chunks.
__device__ __forceinline__ v16bf load_a_frag(const unsigned short* p0,
                                             const unsigned short* p1) {
  union { struct { v8us lo, hi; } s; v16bf b; } t;
  t.s.lo = *(const v8us*)p0;
  t.s.hi = *(const v8us*)(p0 + 8);
  return t.b;
}

// ---- preprocess: f32 weights -> bf16 in workspace ------------------------

__global__ __launch_bounds__(256)
void prep_weights(const float* __restrict__ Wih, const float* __restrict__ Whh,
                  const float* __restrict__ Wcls,
                  unsigned short* __restrict__ wih_b,
                  unsigned short* __restrict__ whh_b,
                  unsigned short* __restrict__ wcls_b) {
  const int NIH = HID * VEC;
  const int NHH = HID * HID;
  const int NCP = NCLSP * HID;
  int i = blockIdx.x * blockDim.x + threadIdx.x;
  if (i < NIH) {
    wih_b[i] = f2b(Wih[i]);
  } else if (i < NIH + NHH) {
    int j = i - NIH;
    whh_b[j] = f2b(Whh[j]);
  } else if (i < NIH + NHH + NCP) {
    int k = i - NIH - NHH;
    int r = k >> 10, c = k & (HID - 1);
    wcls_b[k] = (r < NCLS) ? f2b(Wcls[r * HID + c]) : (unsigned short)0;
  }
}

// ---- fused RNN: embed -> (W_ih GEMM + W_hh GEMM + tanh) x 64 -> classifier

__global__ __launch_bounds__(256)
void rnn_fused(const int* __restrict__ x, const float* __restrict__ emb,
               const unsigned short* __restrict__ Wih,
               const unsigned short* __restrict__ Whh,
               const unsigned short* __restrict__ Wcls,
               const float* __restrict__ bcls, float* __restrict__ out) {
  __shared__ unsigned short xe[BT * VEC];        // 16 KB  (x_t embeddings, bf16)
  __shared__ unsigned short h[2][BT * HID];      // 64 KB  (double-buffered h, bf16)

  const int tid  = threadIdx.x;
  const int lane = tid & 31;
  const int wave = tid >> 5;                     // 8 waves
  const int b0   = blockIdx.x * BT;

  // fragment-layout lane constants (this code's contiguous K order, k = 16h + s)
  const int arow = lane & 15;                    // A: row M
  const int akb  = (lane < 16) ? 0 : 16;         // A: K sub-block base
  const int bcol = lane & 15;                    // B: column N
  const int bkb  = (lane < 16) ? 0 : 16;         // B: K sub-block base

  // h0 = 0
  for (int i = tid; i < BT * HID; i += 256) h[0][i] = 0;
  __syncthreads();

  int cur = 0;
  for (int t = 0; t < SEN; ++t) {
    // ---- gather embeddings for this timestep: xe[16][512] bf16 ----
    {
      int m  = tid >> 4;                 // 0..15 batch row
      int c0 = (tid & 15) * 32;          // 32 cols per thread
      int tok = x[(b0 + m) * SEN + t];
      const float4* er = (const float4*)(emb + tok * VEC + c0);
      uint2* dst = (uint2*)&xe[m * VEC + c0];
      #pragma unroll
      for (int c = 0; c < 8; ++c) {
        float4 v = er[c];                              // global_load_b128
        dst[c] = make_uint2(f2b2(v.x, v.y), f2b2(v.z, v.w));  // ds_store_b64
      }
    }
    __syncthreads();

    // ---- acc = Xe @ W_ih^T + H @ W_hh^T  (per wave: 8 N-tiles) ----
    v8f acc[8];
    #pragma unroll
    for (int j = 0; j < 8; ++j) acc[j] = v8f{0.f,0.f,0.f,0.f,0.f,0.f,0.f,0.f};

    // input-projection phase: K = 512
    for (int k0 = 0; k0 < VEC; k0 += 32) {
      v16bf A = load_a_frag(&xe[arow * VEC + k0 + akb],
                            &xe[arow * VEC + k0 + 16 + akb]);
      #pragma unroll
      for (int j = 0; j < 8; ++j) {
        int n = (wave * 8 + j) * 16 + bcol;
        v16bf B = load_b_frag(&Wih[n * VEC + k0 + bkb]);
        acc[j] = __builtin_amdgcn_wmma_f32_16x16x32_bf16(
            false, A, false, B, (short)0, acc[j], false, false);
      }
    }
    // recurrent phase: K = 1024
    const unsigned short* hc = h[cur];
    for (int k0 = 0; k0 < HID; k0 += 32) {
      v16bf A = load_a_frag(&hc[arow * HID + k0 + akb],
                            &hc[arow * HID + k0 + 16 + akb]);
      #pragma unroll
      for (int j = 0; j < 8; ++j) {
        int n = (wave * 8 + j) * 16 + bcol;
        v16bf B = load_b_frag(&Whh[n * HID + k0 + bkb]);
        acc[j] = __builtin_amdgcn_wmma_f32_16x16x32_bf16(
            false, A, false, B, (short)0, acc[j], false, false);
      }
    }

    // ---- h_new = tanh(acc) -> LDS (next buffer), per D-matrix layout ----
    unsigned short* hn = h[cur ^ 1];
    #pragma unroll
    for (int j = 0; j < 8; ++j) {
      int n = (wave * 8 + j) * 16 + (lane & 15);
      #pragma unroll
      for (int p = 0; p < 8; ++p) {
        int m = p + ((lane >> 4) << 3);          // vgpr p: M=p (lo lanes) / p+8
        hn[m * HID + n] = f2b(fast_tanh(acc[j][p]));
      }
    }
    cur ^= 1;
    __syncthreads();   // publish h_new; fence xe/h reads vs next-step writes
  }

  // ---- classifier: logits[16][65] = h_last @ W_cls^T + b_cls ----
  if (wave < 5) {                                // 5 x 16 = 80 padded cols
    const unsigned short* hc = h[cur];
    v8f acc = v8f{0.f,0.f,0.f,0.f,0.f,0.f,0.f,0.f};
    const int n0 = wave * 16;
    for (int k0 = 0; k0 < HID; k0 += 32) {
      v16bf A = load_a_frag(&hc[arow * HID + k0 + akb],
                            &hc[arow * HID + k0 + 16 + akb]);
      v16bf B = load_b_frag(&Wcls[(n0 + bcol) * HID + k0 + bkb]);
      acc = __builtin_amdgcn_wmma_f32_16x16x32_bf16(
          false, A, false, B, (short)0, acc, false, false);
    }
    int n = n0 + (lane & 15);
    if (n < NCLS) {
      float bias = bcls[n];
      #pragma unroll
      for (int p = 0; p < 8; ++p) {
        int m = p + ((lane >> 4) << 3);
        out[(b0 + m) * NCLS + n] = acc[p] + bias;
      }
    }
  }
}

// ---- host launcher -------------------------------------------------------

extern "C" void kernel_launch(void* const* d_in, const int* in_sizes, int n_in,
                              void* d_out, int out_size, void* d_ws, size_t ws_size,
                              hipStream_t stream) {
  const int*   x     = (const int*)  d_in[0];   // [2048,64] token ids
  const float* emb   = (const float*)d_in[1];   // [29,512]
  const float* W_ih  = (const float*)d_in[2];   // [1024,512]
  const float* W_hh  = (const float*)d_in[3];   // [1024,1024]
  const float* W_cls = (const float*)d_in[4];   // [65,1024]
  const float* b_cls = (const float*)d_in[5];   // [65]
  float* out = (float*)d_out;                   // [2048,65]

  // workspace: bf16 weight copies (W_cls zero-padded to 80 rows), ~3.2 MB
  unsigned short* wih_b  = (unsigned short*)d_ws;
  unsigned short* whh_b  = wih_b + (size_t)HID * VEC;
  unsigned short* wcls_b = whh_b + (size_t)HID * HID;

  const int total = HID * VEC + HID * HID + NCLSP * HID;
  prep_weights<<<(total + 255) / 256, 256, 0, stream>>>(
      W_ih, W_hh, W_cls, wih_b, whh_b, wcls_b);

  rnn_fused<<<BATCH / BT, 256, 0, stream>>>(
      x, emb, wih_b, whh_b, wcls_b, b_cls, out);

  (void)in_sizes; (void)n_in; (void)out_size; (void)ws_size;
}
